// DitTalkingHead_39462159516117
// MI455X (gfx1250) — matrix operand order repaired and checked by reading, not compiled, not run on hardware
//
#include <hip/hip_runtime.h>

// ---------------------------------------------------------------------------
// DitTalkingHead deformable-attention block for MI455X (gfx1250, wave32).
// Heavy GEMMs run on v_wmma_f32_16x16x32_bf16 with a 32x32 per-wave tile and
// register double-buffering; sampling stage is f32 VALU.
// ---------------------------------------------------------------------------

typedef __attribute__((ext_vector_type(16))) __bf16 v16bf;
typedef __attribute__((ext_vector_type(8)))  __bf16 v8bf;
typedef __attribute__((ext_vector_type(8)))  float  v8f;

static constexpr int Bc = 2;
static constexpr int Lc = 4096;
static constexpr int Dc = 1024;
static constexpr int Hc = 16;
static constexpr int Pc = 4;
static constexpr int Dhc = Dc / Hc;       // 64
static constexpr int Mc  = Bc * Lc;       // 8192 rows in every GEMM

// ---------------------------------------------------------------------------
// f32 -> bf16 elementwise conversion (activations)
// ---------------------------------------------------------------------------
__global__ void cvt_bf16_kernel(const float* __restrict__ in,
                                __bf16* __restrict__ out, int n) {
  int i = blockIdx.x * blockDim.x + threadIdx.x;
  if (i < n) out[i] = (__bf16)in[i];
}

// ---------------------------------------------------------------------------
// Pack a K x N (row-major f32) weight matrix into WMMA B-fragment layout.
// Tile (kt, nt) covers K rows [kt*32, kt*32+32), N cols [nt*16, nt*16+16).
// Lane layout (16-bit B, 32x16): lanes 0-15 hold K=0..7,16..23, lanes 16-31
// hold K=8..15,24..31; each lane's 16 bf16 are contiguous in the packed buffer
// so the GEMM reads them with plain b128 loads.
// packed index: ((nt*KT + kt)*32 + lane)*16 + i,
//               K = kt*32 + ((lane&16)>>1) + i + (i&8), N = nt*16 + (lane&15)
// ---------------------------------------------------------------------------
__global__ void pack_w_kernel(const float* __restrict__ W,
                              __bf16* __restrict__ out, int K, int N) {
  int t = blockIdx.x * blockDim.x + threadIdx.x;
  const int KT = K >> 5, NT = N >> 4;
  if (t >= KT * NT * 32) return;
  const int lane = t & 31;
  const int tile = t >> 5;
  const int kt = tile % KT;
  const int nt = tile / KT;
  const int n  = (nt << 4) + (lane & 15);
  const int k0 = (kt << 5) + ((lane & 16) >> 1);
  __bf16* dst = out + (size_t)t * 16;
#pragma unroll
  for (int i = 0; i < 16; ++i)
    dst[i] = (__bf16)W[(size_t)(k0 + i + (i & 8)) * N + n];
}

// ---------------------------------------------------------------------------
// Fragment helpers
// ---------------------------------------------------------------------------
__device__ __forceinline__ v16bf load_afrag(const __bf16* p) {
  // 16-bit A 16x32 layout: lane's halves 0..7 = K(ka..ka+7),
  // halves 8..15 = K(16+ka..16+ka+7): two contiguous 16-byte loads.
  v8bf lo = *(const v8bf*)p;
  v8bf hi = *(const v8bf*)(p + 16);
  v16bf a;
#pragma unroll
  for (int i = 0; i < 8; ++i) { a[i] = lo[i]; a[i + 8] = hi[i]; }
  return a;
}

__device__ __forceinline__ v8f wmma_bf16(v16bf a, v16bf b, v8f c) {
  return __builtin_amdgcn_wmma_f32_16x16x32_bf16(
      /*neg_a=*/false, a, /*neg_b=*/false, b,
      /*c_mod=*/(short)0, c, /*reuse_a=*/false, /*reuse_b=*/false);
}

// ---------------------------------------------------------------------------
// GEMM: C(M x N, f32) = A(M x K, bf16 row-major) * Bpacked + bias
// 32x32 output supertile per wave: 2 A-frags x 2 B-frags -> 4 WMMAs per
// K-step, each fragment reused twice. Register double-buffering: next
// K-step's 8 b128 loads are issued before the current 4 WMMAs so VMEM
// latency overlaps matrix math. Requires M%32==0, N%32==0, K%32==0.
// ---------------------------------------------------------------------------
__global__ void __launch_bounds__(256) gemm_bf16_wmma_kernel(
    const __bf16* __restrict__ A, const __bf16* __restrict__ Bp,
    const float* __restrict__ bias, float* __restrict__ C,
    int M, int N, int K) {
  const int lane = threadIdx.x & 31;
  const int wave = blockIdx.x * (blockDim.x >> 5) + (threadIdx.x >> 5);
  const int NT2 = N >> 5;                 // 32-wide supertiles
  const int MT2 = M >> 5;
  const int KT  = K >> 5;
  if (wave >= MT2 * NT2) return;
  const int mt2 = wave / NT2;             // waves in a block share mt2 -> A reuse
  const int nt2 = wave % NT2;

  const int ka = (lane & 16) >> 1;        // 0 or 8
  const __bf16* arow0 = A + ((size_t)(mt2 << 5) + (lane & 15)) * K + ka;
  const __bf16* arow1 = arow0 + (size_t)16 * K;
  const __bf16* bptr0 = Bp + (size_t)(nt2 * 2) * KT * 512 + (size_t)lane * 16;
  const __bf16* bptr1 = bptr0 + (size_t)KT * 512;

  v8f c00 = {}, c01 = {}, c10 = {}, c11 = {};

  v16bf a0 = load_afrag(arow0);
  v16bf a1 = load_afrag(arow1);
  v16bf b0 = *(const v16bf*)bptr0;
  v16bf b1 = *(const v16bf*)bptr1;

  for (int kt = 0; kt < KT - 1; ++kt) {
    arow0 += 32; arow1 += 32; bptr0 += 512; bptr1 += 512;
    // issue next K-step's loads first so they overlap the 4 WMMAs below
    v16bf a0n = load_afrag(arow0);
    v16bf a1n = load_afrag(arow1);
    v16bf b0n = *(const v16bf*)bptr0;
    v16bf b1n = *(const v16bf*)bptr1;

    c00 = wmma_bf16(a0, b0, c00);
    c01 = wmma_bf16(a0, b1, c01);
    c10 = wmma_bf16(a1, b0, c10);
    c11 = wmma_bf16(a1, b1, c11);

    a0 = a0n; a1 = a1n; b0 = b0n; b1 = b1n;
  }
  c00 = wmma_bf16(a0, b0, c00);
  c01 = wmma_bf16(a0, b1, c01);
  c10 = wmma_bf16(a1, b0, c10);
  c11 = wmma_bf16(a1, b1, c11);

  // C tile layout: VGPR r, lanes 0-15 -> row +r, lanes 16-31 -> row +8+r,
  // column = (lane&15) within each 16-wide tile.
  const int nbase = (nt2 << 5) + (lane & 15);
  const int mbase = (mt2 << 5) + ((lane & 16) >> 1);
  const float bv0 = bias[nbase];
  const float bv1 = bias[nbase + 16];
  float* crow = C + (size_t)mbase * N + nbase;
#pragma unroll
  for (int r = 0; r < 8; ++r) {
    crow[(size_t)r * N]            = c00[r] + bv0;
    crow[(size_t)r * N + 16]       = c01[r] + bv1;
    crow[(size_t)(r + 16) * N]     = c10[r] + bv0;
    crow[(size_t)(r + 16) * N + 16] = c11[r] + bv1;
  }
}

// ---------------------------------------------------------------------------
// Sampler: one wave per (b, l, h). Softmax over P=4 attention logits,
// grid_sample (height==1) bilinear gather along L from vproj (f32), weighted
// sum, write bf16 mix row for the final GEMM. Lane covers 2 of the 64 Dh.
// ---------------------------------------------------------------------------
__global__ void __launch_bounds__(256) sampler_kernel(
    const float* __restrict__ vproj,   // (B*L, D)  f32  (v projection)
    const float* __restrict__ offb,    // (B*L, 128) f32 (H,P,2 offsets)
    const float* __restrict__ awb,     // (B*L, 64)  f32 (H,P logits)
    __bf16* __restrict__ mixbf) {      // (B*L, D)  bf16
  const int lane = threadIdx.x & 31;
  const int wave = blockIdx.x * (blockDim.x >> 5) + (threadIdx.x >> 5);
  if (wave >= Bc * Lc * Hc) return;
  const int h  = wave % Hc;
  const int bl = wave / Hc;            // b*L + l
  const int l  = bl % Lc;
  const int b  = bl / Lc;

  // softmax over P=4 logits
  const float* aw = awb + (size_t)bl * (Hc * Pc) + h * Pc;
  float a0 = aw[0], a1 = aw[1], a2 = aw[2], a3 = aw[3];
  float mx = fmaxf(fmaxf(a0, a1), fmaxf(a2, a3));
  float e0 = expf(a0 - mx), e1 = expf(a1 - mx);
  float e2 = expf(a2 - mx), e3 = expf(a3 - mx);
  float inv = 1.0f / (e0 + e1 + e2 + e3);
  float wp[4] = {e0 * inv, e1 * inv, e2 * inv, e3 * inv};

  const float refy = (float)l * (1.0f / (float)(Lc - 1));
  const float* off = offb + (size_t)bl * (Hc * Pc * 2) + h * (Pc * 2);
  const float* vb  = vproj + (size_t)b * Lc * Dc + h * Dhc + lane * 2;

  float acc0 = 0.0f, acc1 = 0.0f;
#pragma unroll
  for (int p = 0; p < Pc; ++p) {
    float sx = fminf(fmaxf(off[p * 2 + 0], 0.0f), 1.0f);
    float sy = fminf(fmaxf(refy + off[p * 2 + 1], 0.0f), 1.0f);
    float ix = ((sx + 1.0f) * (float)Lc - 1.0f) * 0.5f;
    float iy = sy * 0.5f;                       // ((sy+1)*1 - 1)*0.5
    float x0 = floorf(ix);
    float fx = ix - x0;
    int   x0i = (int)x0;
    int   x1i = x0i + 1;
    float m0 = (x0i >= 0 && x0i < Lc) ? 1.0f : 0.0f;
    float m1 = (x1i >= 0 && x1i < Lc) ? 1.0f : 0.0f;
    float y0 = floorf(iy);
    float fy = iy - y0;
    float hy = ((y0 == 0.0f) ? (1.0f - fy) : 0.0f) +
               ((y0 == -1.0f) ? fy : 0.0f);
    float w0 = (1.0f - fx) * m0 * hy * wp[p];
    float w1 = fx * m1 * hy * wp[p];
    int c0 = x0i < 0 ? 0 : (x0i > Lc - 1 ? Lc - 1 : x0i);
    int c1 = x1i < 0 ? 0 : (x1i > Lc - 1 ? Lc - 1 : x1i);
    const float* r0 = vb + (size_t)c0 * Dc;
    const float* r1 = vb + (size_t)c1 * Dc;
    acc0 += w0 * r0[0] + w1 * r1[0];
    acc1 += w0 * r0[1] + w1 * r1[1];
  }
  __bf16* o = mixbf + (size_t)bl * Dc + h * Dhc + lane * 2;
  o[0] = (__bf16)acc0;
  o[1] = (__bf16)acc1;
}

// ---------------------------------------------------------------------------
// Launch
// ---------------------------------------------------------------------------
extern "C" void kernel_launch(void* const* d_in, const int* in_sizes, int n_in,
                              void* d_out, int out_size, void* d_ws, size_t ws_size,
                              hipStream_t stream) {
  (void)in_sizes; (void)n_in; (void)out_size; (void)ws_size;

  const float* query   = (const float*)d_in[0];
  // d_in[1] = key  (unused by reference math path)
  const float* value   = (const float*)d_in[2];
  // d_in[3]/d_in[4] = w_qk/b_qk (dead code in reference)
  const float* w_value = (const float*)d_in[5];
  const float* b_value = (const float*)d_in[6];
  const float* w_off   = (const float*)d_in[7];
  const float* b_off   = (const float*)d_in[8];
  const float* w_attw  = (const float*)d_in[9];
  const float* b_attw  = (const float*)d_in[10];
  const float* w_out   = (const float*)d_in[11];
  const float* b_out   = (const float*)d_in[12];
  float* out = (float*)d_out;

  // workspace layout (bytes)
  char* ws = (char*)d_ws;
  __bf16* q_bf     = (__bf16*)(ws + 0);          // 16 MiB
  __bf16* v_bf     = (__bf16*)(ws + 16777216);   // 16 MiB
  __bf16* mix_bf   = (__bf16*)(ws + 33554432);   // 16 MiB
  __bf16* wv_pack  = (__bf16*)(ws + 50331648);   //  2 MiB
  __bf16* wo_pack  = (__bf16*)(ws + 52428800);   //  2 MiB
  __bf16* wof_pack = (__bf16*)(ws + 54525952);   // 256 KiB
  __bf16* wat_pack = (__bf16*)(ws + 54788096);   // 128 KiB
  float*  vproj    = (float*) (ws + 54919168);   // 32 MiB
  float*  offbuf   = (float*) (ws + 88473600);   //  4 MiB
  float*  attwbuf  = (float*) (ws + 92667904);   //  2 MiB
  // total ~ 90.4 MiB

  const int nElem = Mc * Dc;                      // 8,388,608

  // 1) activations -> bf16
  cvt_bf16_kernel<<<(nElem + 255) / 256, 256, 0, stream>>>(query, q_bf, nElem);
  cvt_bf16_kernel<<<(nElem + 255) / 256, 256, 0, stream>>>(value, v_bf, nElem);

  // 2) weight packs (threads = KT*NT*32)
  auto packThreads = [](int K, int N) { return (K >> 5) * (N >> 4) * 32; };
  {
    int t = packThreads(Dc, Dc);
    pack_w_kernel<<<(t + 255) / 256, 256, 0, stream>>>(w_value, wv_pack, Dc, Dc);
    pack_w_kernel<<<(t + 255) / 256, 256, 0, stream>>>(w_out,   wo_pack, Dc, Dc);
  }
  {
    int t = packThreads(Dc, Hc * Pc * 2);         // N=128
    pack_w_kernel<<<(t + 255) / 256, 256, 0, stream>>>(w_off, wof_pack, Dc, Hc * Pc * 2);
  }
  {
    int t = packThreads(Dc, Hc * Pc);             // N=64
    pack_w_kernel<<<(t + 255) / 256, 256, 0, stream>>>(w_attw, wat_pack, Dc, Hc * Pc);
  }

  // 3) GEMMs (one 32x32 supertile per wave, 8 waves/block)
  auto gemmBlocks = [](int M, int N) {
    int waves = (M >> 5) * (N >> 5);
    return (waves + 7) / 8;
  };
  // v projection: (B*L, D) = value_bf @ w_value + b_value
  gemm_bf16_wmma_kernel<<<gemmBlocks(Mc, Dc), 256, 0, stream>>>(
      v_bf, wv_pack, b_value, vproj, Mc, Dc, Dc);
  // offsets: (B*L, 128) = query_bf @ w_off + b_off
  gemm_bf16_wmma_kernel<<<gemmBlocks(Mc, Hc * Pc * 2), 256, 0, stream>>>(
      q_bf, wof_pack, b_off, offbuf, Mc, Hc * Pc * 2, Dc);
  // attention logits: (B*L, 64) = query_bf @ w_attw + b_attw
  gemm_bf16_wmma_kernel<<<gemmBlocks(Mc, Hc * Pc), 256, 0, stream>>>(
      q_bf, wat_pack, b_attw, attwbuf, Mc, Hc * Pc, Dc);

  // 4) softmax + grid-sample gather + head mix -> bf16
  {
    int waves = Bc * Lc * Hc;                     // 131072
    sampler_kernel<<<(waves + 7) / 8, 256, 0, stream>>>(
        vproj, offbuf, attwbuf, mix_bf);
  }

  // 5) output projection: out = mix_bf @ w_out + b_out   (f32 result)
  gemm_bf16_wmma_kernel<<<gemmBlocks(Mc, Dc), 256, 0, stream>>>(
      mix_bf, wo_pack, b_out, out, Mc, Dc, Dc);
}
